// Prototypical_75754633167403
// MI455X (gfx1250) — compile-verified, matrix operand order checked by reading, not run on hardware
//
#include <hip/hip_runtime.h>
#include <hip/hip_bf16.h>

typedef _Float16 half_t;
typedef __attribute__((ext_vector_type(16))) _Float16 v16h;
typedef __attribute__((ext_vector_type(8)))  float    v8f;

#define IMW 84
#define IMH 84
#define WS84 96       // padded row stride at 84-res (6 tiles of 16)
#define WS42 48       // padded row stride at 42-res (3 tiles of 16)
#define D_FEAT 7056   // 21*21*16 (final features are compact)
#define N1 640        // pass1: 160 support + 320 query + 160 flipped support
#define N2 480        // pass2: 160 support + 320 query

// Flip to 0 if the assembler rejects the async mnemonic.
#ifndef USE_ASYNC_LDS
#define USE_ASYNC_LDS 1
#endif

__device__ __forceinline__ void wait_async_lds() {
#if __has_builtin(__builtin_amdgcn_s_wait_asynccnt)
    __builtin_amdgcn_s_wait_asynccnt(0);
#else
    asm volatile("s_wait_asynccnt 0x0" ::: "memory");
#endif
}

// ---------------------------------------------------------------------------
// Weight pre-swizzle: (3,3,Cin,16) f32 -> wave32 WMMA-B fragments, fp16.
// Per layer: 5 K-pairs (taps 0..8, tap 9 zero) x 32 lanes x 16 halves.
// Lane L holds col N=L&15; elems 0..7 = K[(L>>4)*8 .. +7], elems 8..15 = K+16.
// K within a pair: K0..15 = tapA channels, K16..31 = tapB channels.
// ---------------------------------------------------------------------------
__global__ void prep_weights_kernel(const float* k1, const float* k2, const float* k3,
                                    const float* k4, const float* k5, const float* k6,
                                    const float* k7, const float* k8, half_t* wbuf) {
    const float* ks[8] = {k1, k2, k3, k4, k5, k6, k7, k8};
    int idx = blockIdx.x * blockDim.x + threadIdx.x;
    if (idx >= 8 * 5 * 32 * 16) return;
    int l   = idx / 2560;
    int rem = idx - l * 2560;
    int p   = rem >> 9;          // pair 0..4
    int L   = (rem >> 4) & 31;   // lane
    int e   = rem & 15;          // element
    int k_local = (e & 7) | ((L >> 4) << 3) | ((e >> 3) << 4);  // 0..31
    int tap = p * 2 + (k_local >> 4);
    int cin = k_local & 15;
    int n   = L & 15;
    int Cin = (l == 0) ? 3 : 16;
    float v = 0.0f;
    if (tap < 9 && cin < Cin) v = ks[l][(tap * Cin + cin) * 16 + n];
    wbuf[idx] = (half_t)v;
}

// ---------------------------------------------------------------------------
// Image prep: f32 NHWC(C=3) -> fp16 NHWC(C=16, zero pad) with padded row
// stride WS84 (x in [84,96) zeroed). flip=1 appends axis(-2)-flipped support.
// ---------------------------------------------------------------------------
__global__ void prep_input_kernel(const float* __restrict__ support,
                                  const float* __restrict__ query,
                                  half_t* __restrict__ out, int nTotal, int withFlip) {
    long long idx = (long long)blockIdx.x * blockDim.x + threadIdx.x;
    long long total = (long long)nTotal * IMW * WS84 * 16;
    if (idx >= total) return;
    int ch = (int)(idx & 15);
    long long pix = idx >> 4;
    int x = (int)(pix % WS84); pix /= WS84;
    int y = (int)(pix % IMW); pix /= IMW;
    int n = (int)pix;
    float v = 0.0f;
    if (ch < 3 && x < IMH) {
        if (n < 160) {
            v = support[(((long long)n * IMW + y) * IMH + x) * 3 + ch];
        } else if (n < 480) {
            v = query[((((long long)n - 160) * IMW + y) * IMH + x) * 3 + ch];
        } else if (withFlip) {  // flipped support: flip axis=-2 (second spatial dim)
            v = support[((((long long)n - 480) * IMW + y) * IMH + (IMH - 1 - x)) * 3 + ch];
        }
    }
    out[idx] = (half_t)v;
}

// ---------------------------------------------------------------------------
// 3x3 SAME conv, 16->16 channels, via v_wmma_f32_16x16x32_f16.
// Activations use padded row stride WS (= #tiles*16) so the store epilogue is
// fully unconditional. Block = ceil(W/16) waves x R output rows of one image.
// R+2 input rows staged once in LDS (async global->LDS) with 1-column zero
// halo (padded width WP) so the inner loop is ds_load_b128 + wmma only.
// B fragments preloaded into registers. out = scale*(acc + bias), fp16.
// ---------------------------------------------------------------------------
__global__ void conv3x3_wmma_kernel(const half_t* __restrict__ in,
                                    half_t* __restrict__ out,
                                    const half_t* __restrict__ wb,   // [5][32][16]
                                    const float* __restrict__ bias,
                                    const float* __restrict__ wm, int wmIdx,
                                    int H, int W, int WS, int R, int WP) {
    extern __shared__ char smem[];
    half_t* srow = (half_t*)smem;                 // [(R+2)][WP][16]
    const int img = blockIdx.y;
    const int y0  = blockIdx.x * R;
    const int tid = threadIdx.x;
    const long long imgStride = (long long)H * WS * 16;
    const half_t* ibase = in + (long long)img * imgStride;

    // ---- Stage R+2 rows (halo columns + OOB rows/cols zeroed) ----
    const int rows = R + 2;
    const int chunksPerRow = WP * 2;              // 16-byte chunks per padded row
    for (int i = tid; i < rows * chunksPerRow; i += blockDim.x) {
        int rr  = i / chunksPerRow;
        int o   = i - rr * chunksPerRow;
        int col = o >> 1;                          // padded column
        int sx  = col - 1;
        int yy  = y0 - 1 + rr;
        half_t* dst = srow + ((rr * WP + col) << 4) + ((o & 1) << 3);
        if (yy >= 0 && yy < H && sx >= 0 && sx < W) {
            const half_t* src = ibase + (((long long)yy * WS + sx) << 4) + ((o & 1) << 3);
#if USE_ASYNC_LDS
            unsigned int loff = (unsigned int)(uintptr_t)dst;          // LDS byte addr
            asm volatile("global_load_async_to_lds_b128 %0, %1, off"
                         :: "v"(loff), "v"((unsigned long long)(uintptr_t)src)
                         : "memory");
#else
            *(int4*)dst = *(const int4*)src;
#endif
        } else {
            int4 z; z.x = 0; z.y = 0; z.z = 0; z.w = 0;
            *(int4*)dst = z;
        }
    }
#if USE_ASYNC_LDS
    wait_async_lds();
#endif
    __syncthreads();

    const int wave = tid >> 5, lane = tid & 31;
    const int x0   = wave * 16;
    const int pcol = x0 + (lane & 15);   // A-row pixel; padded col = pcol + tap%3
    const int chb  = (lane >> 4) << 3;   // channel sub-block 0 or 8

    // ---- Preload the 5 B fragments (shared across all R rows) ----
    v16h bfr[5];
#pragma unroll
    for (int p = 0; p < 5; ++p)
        bfr[p] = *(const v16h*)(wb + (p * 32 + lane) * 16);

    const float sc = (wmIdx < 0) ? 1.0f : wm[wmIdx];
    const int n = lane & 15;
    const int mbase = (lane >> 4) * 8;
    const float bn = bias[n];

    for (int yl = 0; yl < R; ++yl) {
        v8f acc = {};
#pragma unroll
        for (int p = 0; p < 5; ++p) {
            union { v16h v; int4 q[2]; } a;
#pragma unroll
            for (int h = 0; h < 2; ++h) {
                int tap = p * 2 + h;
                if (tap < 9) {
                    int dy = tap / 3, dxc = tap % 3;
                    a.q[h] = *(const int4*)(srow +
                        ((((yl + dy) * WP + pcol + dxc)) << 4) + chb);
                } else {
                    int4 z; z.x = 0; z.y = 0; z.z = 0; z.w = 0;
                    a.q[h] = z;
                }
            }
            acc = __builtin_amdgcn_wmma_f32_16x16x32_f16(
                false, a.v, false, bfr[p], (short)0, acc, false, false);
        }
        // D layout: lane holds col n=lane&15; VGPR r -> row M = r + 8*(lane>>4).
        // Padded stride => every store is in-bounds: no exec-mask chain.
        half_t* obase = out + (long long)img * imgStride + (long long)(y0 + yl) * WS * 16;
#pragma unroll
        for (int r = 0; r < 8; ++r) {
            int pm = x0 + mbase + r;
            obase[pm * 16 + n] = (half_t)(sc * (acc[r] + bn));
        }
    }
}

// ---------------------------------------------------------------------------
// Fused: v = sa*(g*x+be); v = sb*relu(v); m = 2x2 maxpool(v); out = sc*sd*m.
// Reads padded stride WSin. Writes fp16 with padded stride WSout (mid-net) or
// compact f32 flattened features (final).
// ---------------------------------------------------------------------------
__device__ __forceinline__ float wm_get(const float* wm, int i) {
    return (i < 0) ? 1.0f : wm[i];
}

__global__ void scale_relu_pool_kernel(const half_t* __restrict__ in,
                                       half_t* __restrict__ outh,
                                       float* __restrict__ outf,
                                       const float* __restrict__ g,
                                       const float* __restrict__ be,
                                       const float* __restrict__ wm,
                                       int ia, int ib, int ic, int id,
                                       int H, int W, int WSin, int WSout,
                                       long long total) {
    long long idx = (long long)blockIdx.x * blockDim.x + threadIdx.x;
    if (idx >= total) return;
    const int oW = W >> 1, oH = H >> 1;
    int c = (int)(idx & 15);
    long long p = idx >> 4;
    int ox = (int)(p % oW); p /= oW;
    int oy = (int)(p % oH); p /= oH;
    long long nimg = p;
    const float sa = wm_get(wm, ia), sb = wm_get(wm, ib);
    const float sc = wm_get(wm, ic) * wm_get(wm, id);
    const float gc = g[c], bec = be[c];
    const half_t* base = in + ((nimg * H + 2 * oy) * WSin + 2 * ox) * 16 + c;
    float m = -3.4e38f;
#pragma unroll
    for (int j = 0; j < 4; ++j) {
        float x = (float)base[(j >> 1) * WSin * 16 + (j & 1) * 16];
        float v = sa * (gc * x + bec);
        v = fmaxf(v, 0.0f) * sb;
        m = fmaxf(m, v);
    }
    float r = m * sc;
    if (outf) outf[idx] = r;                                  // compact flatten
    else      outh[((nimg * oH + oy) * WSout + ox) * 16 + c] = (half_t)r;
}

// ---------------------------------------------------------------------------
// Class prototypes: mean over 8 support embeddings.
// ---------------------------------------------------------------------------
__global__ void protos_kernel(const float* __restrict__ zp, float* __restrict__ protos) {
    int idx = blockIdx.x * blockDim.x + threadIdx.x;
    if (idx >= 20 * D_FEAT) return;
    int c = idx / D_FEAT, d = idx - c * D_FEAT;
    float s = 0.0f;
#pragma unroll
    for (int j = 0; j < 8; ++j) s += zp[(long long)(c * 8 + j) * D_FEAT + d];
    protos[idx] = s * 0.125f;
}

// ---------------------------------------------------------------------------
// Mean-squared-distance between selected row pairs; one block per pair.
// mode 0: aug (i,i); 1: intra-class; 2: vs next class; 3: query x proto.
// ---------------------------------------------------------------------------
__global__ void pair_msq_kernel(const float* __restrict__ A, const float* __restrict__ B,
                                float* __restrict__ out, int mode) {
    __shared__ float red[256];
    int i = blockIdx.x;
    int ra, rb;
    if (mode == 0)      { ra = i; rb = i; }
    else if (mode == 1) { int c = i >> 6, s = (i >> 3) & 7, s2 = i & 7;
                          ra = c * 8 + s; rb = c * 8 + s2; }
    else if (mode == 2) { int c = i >> 6, s = (i >> 3) & 7, s2 = i & 7;
                          ra = c * 8 + s; rb = ((c + 1) % 20) * 8 + s2; }
    else                { ra = i / 20; rb = i % 20; }
    const float* a = A + (long long)ra * D_FEAT;
    const float* b = B + (long long)rb * D_FEAT;
    float sum = 0.0f;
    for (int d = threadIdx.x; d < D_FEAT; d += blockDim.x) {
        float t = a[d] - b[d];
        sum += t * t;
    }
    red[threadIdx.x] = sum;
    __syncthreads();
    for (int st = 128; st > 0; st >>= 1) {
        if (threadIdx.x < st) red[threadIdx.x] += red[threadIdx.x + st];
        __syncthreads();
    }
    if (threadIdx.x == 0) out[i] = red[0] * (1.0f / D_FEAT);
}

// ---------------------------------------------------------------------------
// uns_loss = sum over (c,s) of (d_aug^2 + sum_s2 pos^2 - sum_s2 neg^2)/17 + 0.5
// ---------------------------------------------------------------------------
__global__ void uns_kernel(const float* __restrict__ maug, const float* __restrict__ mpos,
                           const float* __restrict__ mneg, float* __restrict__ uns) {
    __shared__ float red[256];
    int t = threadIdx.x;
    float v = 0.0f;
    if (t < 160) {
        float da = maug[t] * maug[t];
        float pos = 0.0f, neg = 0.0f;
#pragma unroll
        for (int s2 = 0; s2 < 8; ++s2) {
            float p = mpos[t * 8 + s2]; pos += p * p;
            float q = mneg[t * 8 + s2]; neg += q * q;
        }
        v = (da + pos - neg) * (1.0f / 17.0f) + 0.5f;
    }
    red[t] = v;
    __syncthreads();
    for (int st = 128; st > 0; st >>= 1) {
        if (t < st) red[t] += red[t + st];
        __syncthreads();
    }
    if (t == 0) uns[0] = red[0];
}

// ---------------------------------------------------------------------------
// CE + accuracy over 320 query rows x 20 classes; writes final (loss, acc).
// ---------------------------------------------------------------------------
__global__ void ce_kernel(const float* __restrict__ dists, const float* __restrict__ uns,
                          float* __restrict__ outp) {
    __shared__ float rc[384];
    __shared__ float ra[384];
    int t = threadIdx.x;
    float ce = 0.0f, ac = 0.0f;
    if (t < 320) {
        const float* d = dists + t * 20;
        float dmin = d[0]; int am = 0;
        for (int j = 1; j < 20; ++j) if (d[j] < dmin) { dmin = d[j]; am = j; }
        float se = 0.0f;
        for (int j = 0; j < 20; ++j) se += __expf(dmin - d[j]);
        int y = t / 16;                       // queries ordered (class, within-class)
        ce = d[y] - dmin + __logf(se);        // -log_softmax(-d)[y]
        ac = (am == y) ? 1.0f : 0.0f;
    }
    rc[t] = ce; ra[t] = ac;
    __syncthreads();
    for (int st = 192; st > 0; st >>= 1) {
        if (t < st) { rc[t] += rc[t + st]; ra[t] += ra[t + st]; }
        __syncthreads();
    }
    if (t == 0) {
        outp[0] = rc[0] * (1.0f / 320.0f) + uns[0];
        outp[1] = ra[0] * (1.0f / 320.0f);
    }
}

// ---------------------------------------------------------------------------
// Host orchestration
// ---------------------------------------------------------------------------
extern "C" void kernel_launch(void* const* d_in, const int* in_sizes, int n_in,
                              void* d_out, int out_size, void* d_ws, size_t ws_size,
                              hipStream_t stream) {
    const float* support = (const float*)d_in[0];
    const float* query   = (const float*)d_in[1];
    const float* kk[8];  for (int i = 0; i < 8; ++i) kk[i] = (const float*)d_in[2 + i];
    const float* bb[8];  for (int i = 0; i < 8; ++i) bb[i] = (const float*)d_in[10 + i];
    const float* g1  = (const float*)d_in[18];
    const float* be1 = (const float*)d_in[19];
    const float* g2  = (const float*)d_in[20];
    const float* be2 = (const float*)d_in[21];
    const float* Wm  = (const float*)d_in[22];
    float* outp = (float*)d_out;

    // Workspace carve-up (256B aligned). Ping-pong buffers sized for the
    // padded 84-res layout (96-wide), which also covers the 42-res layers.
    char* w = (char*)d_ws;
    auto carve = [&](size_t bytes) { char* p = w; w += (bytes + 255) & ~(size_t)255; return p; };
    half_t* pp0    = (half_t*)carve((size_t)N1 * IMW * WS84 * 16 * sizeof(half_t));
    half_t* pp1    = (half_t*)carve((size_t)N1 * IMW * WS84 * 16 * sizeof(half_t));
    float*  zfeat  = (float*) carve((size_t)N1 * D_FEAT * sizeof(float));
    float*  zp     = (float*) carve((size_t)N2 * D_FEAT * sizeof(float));
    half_t* wbuf   = (half_t*)carve(8 * 5 * 32 * 16 * sizeof(half_t));
    float*  protos = (float*) carve(20 * D_FEAT * sizeof(float));
    float*  maug   = (float*) carve(160 * sizeof(float));
    float*  mpos   = (float*) carve(1280 * sizeof(float));
    float*  mneg   = (float*) carve(1280 * sizeof(float));
    float*  dq     = (float*) carve(6400 * sizeof(float));
    float*  unsb   = (float*) carve(sizeof(float));

    prep_weights_kernel<<<80, 256, 0, stream>>>(kk[0], kk[1], kk[2], kk[3],
                                                kk[4], kk[5], kk[6], kk[7], wbuf);

    auto conv = [&](const half_t* src, half_t* dst, int layer, int wmIdx,
                    int H, int W, int N) {
        int waves = (W + 15) / 16;
        int WSl = waves * 16;                       // padded row stride
        int R  = (H == 84) ? 12 : 14;               // 84 = 7*12, 42 = 3*14
        int WP = WSl + 2;                           // halo-padded LDS width
        dim3 grid(H / R, N);
        size_t lds = (size_t)(R + 2) * WP * 16 * sizeof(half_t);
        conv3x3_wmma_kernel<<<grid, waves * 32, lds, stream>>>(
            src, dst, wbuf + (size_t)layer * 5 * 32 * 16, bb[layer], Wm, wmIdx,
            H, W, WSl, R, WP);
    };
    auto pool = [&](const half_t* src, half_t* dsth, float* dstf,
                    const float* g, const float* be,
                    int ia, int ib, int ic, int id,
                    int H, int W, int WSin, int WSout, int N) {
        long long total = (long long)N * (H / 2) * (W / 2) * 16;
        int blocks = (int)((total + 255) / 256);
        scale_relu_pool_kernel<<<blocks, 256, 0, stream>>>(
            src, dsth, dstf, g, be, Wm, ia, ib, ic, id, H, W, WSin, WSout, total);
    };

    // ---- Pass 1: plain encoder over 640 images (cat + flipped support) ----
    {
        long long total = (long long)N1 * IMW * WS84 * 16;
        prep_input_kernel<<<(int)((total + 255) / 256), 256, 0, stream>>>(
            support, query, pp0, N1, 1);
    }
    conv(pp0, pp1, 0, -1, 84, 84, N1);
    conv(pp1, pp0, 1, -1, 84, 84, N1);
    conv(pp0, pp1, 2, -1, 84, 84, N1);
    conv(pp1, pp0, 3, -1, 84, 84, N1);
    pool(pp0, pp1, nullptr, g1, be1, -1, -1, -1, -1, 84, 84, WS84, WS42, N1);
    conv(pp1, pp0, 4, -1, 42, 42, N1);
    conv(pp0, pp1, 5, -1, 42, 42, N1);
    conv(pp1, pp0, 6, -1, 42, 42, N1);
    conv(pp0, pp1, 7, -1, 42, 42, N1);
    pool(pp1, nullptr, zfeat, g2, be2, -1, -1, -1, -1, 42, 42, WS42, 0, N1);

    // ---- Pass 2: pro encoder (Wm-scaled) over 480 images ----
    {
        long long total = (long long)N2 * IMW * WS84 * 16;
        prep_input_kernel<<<(int)((total + 255) / 256), 256, 0, stream>>>(
            support, query, pp0, N2, 0);
    }
    conv(pp0, pp1, 0, 0, 84, 84, N2);
    conv(pp1, pp0, 1, 1, 84, 84, N2);
    conv(pp0, pp1, 2, 2, 84, 84, N2);
    conv(pp1, pp0, 3, 3, 84, 84, N2);
    pool(pp0, pp1, nullptr, g1, be1, 4, 5, 6, -1, 84, 84, WS84, WS42, N2);
    conv(pp1, pp0, 4, 7, 42, 42, N2);
    conv(pp0, pp1, 5, 8, 42, 42, N2);
    conv(pp1, pp0, 6, 9, 42, 42, N2);
    conv(pp0, pp1, 7, 10, 42, 42, N2);
    pool(pp1, nullptr, zp, g2, be2, 11, 12, 13, 14, 42, 42, WS42, 0, N2);

    // ---- Loss tail ----
    protos_kernel<<<(20 * D_FEAT + 255) / 256, 256, 0, stream>>>(zp, protos);
    pair_msq_kernel<<<160,  256, 0, stream>>>(zfeat, zfeat + (long long)480 * D_FEAT, maug, 0);
    pair_msq_kernel<<<1280, 256, 0, stream>>>(zfeat, zfeat, mpos, 1);
    pair_msq_kernel<<<1280, 256, 0, stream>>>(zfeat, zfeat, mneg, 2);
    pair_msq_kernel<<<6400, 256, 0, stream>>>(zp + (long long)160 * D_FEAT, protos, dq, 3);
    uns_kernel<<<1, 256, 0, stream>>>(maug, mpos, mneg, unsb);
    ce_kernel<<<1, 384, 0, stream>>>(dq, unsb, outp);
}
